// MLAAttention_72327249264724
// MI455X (gfx1250) — matrix operand order repaired and checked by reading, not compiled, not run on hardware
//
#include <hip/hip_runtime.h>
#include <hip/hip_bf16.h>

// ---------------------------------------------------------------------------
// MLA attention for gfx1250: bf16 WMMA GEMMs + flash-attention.
// ---------------------------------------------------------------------------

typedef __attribute__((ext_vector_type(16))) __bf16 v16bf;
typedef __attribute__((ext_vector_type(8)))  __bf16 v8bf;
typedef __attribute__((ext_vector_type(4)))  __bf16 v4bf;
typedef __attribute__((ext_vector_type(8)))  float  v8f;
typedef __attribute__((ext_vector_type(4)))  float  v4f;

#define WMMA_BF16(A, Bm, C) \
  __builtin_amdgcn_wmma_f32_16x16x32_bf16(false, (A), false, (Bm), (short)0, (C), false, false)

constexpr int B_  = 2;
constexpr int S_  = 2048;
constexpr int H_  = 4096;
constexpr int NH_ = 32;
constexpr int HD_ = 128;
constexpr int LAT_ = 1024;
constexpr int TOK_ = B_ * S_;           // 4096 tokens

__device__ inline v16bf combine16(v8bf lo, v8bf hi) {
  v16bf r;
#pragma unroll
  for (int i = 0; i < 8; ++i) { r[i] = lo[i]; r[i + 8] = hi[i]; }
  return r;
}

// ---------------------------------------------------------------------------
// fp32 -> bf16 elementwise convert (vectorized x4)
// ---------------------------------------------------------------------------
__global__ void cvt_f32_bf16(const float* __restrict__ src,
                             __bf16* __restrict__ dst, long n4) {
  long i = (long)blockIdx.x * blockDim.x + threadIdx.x;
  long stride = (long)gridDim.x * blockDim.x;
  const v4f* s4 = (const v4f*)src;
  v4bf* d4 = (v4bf*)dst;
  for (; i < n4; i += stride) {
    v4f x = s4[i];
    v4bf y;
    y[0] = (__bf16)x[0]; y[1] = (__bf16)x[1];
    y[2] = (__bf16)x[2]; y[3] = (__bf16)x[3];
    d4[i] = y;
  }
}

// ---------------------------------------------------------------------------
// GEMM: C[M,N] = A[M,K](bf16,row-major) * W[N,K](bf16,row-major)^T
// Each wave computes a 32x64 tile (2x4 grid of 16x16 WMMA accum).
// OUT_MODE: 0 = fp32 row-major, 1 = bf16 row-major,
//           2 = bf16 "V-transposed" store: [B,NH,HD,S]
// ---------------------------------------------------------------------------
template <int OUT_MODE>
__global__ __launch_bounds__(256) void gemm_bf16(
    const __bf16* __restrict__ A, const __bf16* __restrict__ W,
    void* __restrict__ C, int M, int N, int K) {
  const int lane = threadIdx.x & 31;
  const int wave = blockIdx.x * (blockDim.x >> 5) + (threadIdx.x >> 5);
  const int tilesN = N >> 6;
  const int m0 = (wave / tilesN) * 32;
  const int n0 = (wave % tilesN) * 64;
  if (m0 >= M) return;

  const int half = lane >> 4;
  const int l16  = lane & 15;

  v8f acc[2][4];
#pragma unroll
  for (int mt = 0; mt < 2; ++mt)
#pragma unroll
    for (int nt = 0; nt < 4; ++nt)
#pragma unroll
      for (int j = 0; j < 8; ++j) acc[mt][nt][j] = 0.0f;

  for (int k0 = 0; k0 < K; k0 += 32) {
    // A fragments: lane l16 owns row; K chunks {0..7,16..23}/{8..15,24..31}
    v16bf afr[2];
#pragma unroll
    for (int mt = 0; mt < 2; ++mt) {
      const __bf16* p = A + (long)(m0 + mt * 16 + l16) * K + k0 + 8 * half;
      afr[mt] = combine16(*(const v8bf*)p, *(const v8bf*)(p + 16));
      __builtin_prefetch((const void*)(p + 32), 0, 1);
    }
#pragma unroll
    for (int nt = 0; nt < 4; ++nt) {
      // B fragment: lane l16 owns column n (= row of W); K chunks {0..7,16..23}/{8..15,24..31}
      const __bf16* p = W + (long)(n0 + nt * 16 + l16) * K + k0 + 8 * half;
      v16bf bfr = combine16(*(const v8bf*)p, *(const v8bf*)(p + 16));
      __builtin_prefetch((const void*)(p + 32), 0, 1);
      acc[0][nt] = WMMA_BF16(afr[0], bfr, acc[0][nt]);
      acc[1][nt] = WMMA_BF16(afr[1], bfr, acc[1][nt]);
    }
  }

#pragma unroll
  for (int mt = 0; mt < 2; ++mt)
#pragma unroll
    for (int nt = 0; nt < 4; ++nt)
#pragma unroll
      for (int j = 0; j < 8; ++j) {
        int row = m0 + mt * 16 + j + 8 * half;
        int col = n0 + nt * 16 + l16;
        float v = acc[mt][nt][j];
        if (OUT_MODE == 0) {
          ((float*)C)[(long)row * N + col] = v;
        } else if (OUT_MODE == 1) {
          ((__bf16*)C)[(long)row * N + col] = (__bf16)v;
        } else {
          // row = b*S+s token; col = h*HD+d  ->  [b][h][d][s]
          int b = row >> 11, s = row & (S_ - 1);
          int h = col >> 7,  d = col & (HD_ - 1);
          ((__bf16*)C)[(((long)(b * NH_ + h) * HD_ + d) << 11) + s] = (__bf16)v;
        }
      }
}

// ---------------------------------------------------------------------------
// RoPE: x[B,S,NH,HD] bf16 -> out bf16;  out = (x*cos + rotate_half(x)*sin)*scale
// ---------------------------------------------------------------------------
__global__ void rope_kernel(const __bf16* __restrict__ x,
                            const float* __restrict__ cosp,
                            const float* __restrict__ sinp,
                            __bf16* __restrict__ out, float scale) {
  long n = (long)TOK_ * H_;
  long i = (long)blockIdx.x * blockDim.x + threadIdx.x;
  long stride = (long)gridDim.x * blockDim.x;
  for (; i < n; i += stride) {
    int d = (int)(i & (HD_ - 1));
    long tok = i >> 12;                 // (b*S+s), since NH*HD = 4096
    int s = (int)(tok & (S_ - 1));
    float c  = cosp[s * HD_ + d];
    float sn = sinp[s * HD_ + d];
    float xv = (float)x[i];
    float ot = (d < 64) ? -(float)x[i + 64] : (float)x[i - 64];
    out[i] = (__bf16)((xv * c + ot * sn) * scale);
  }
}

// ---------------------------------------------------------------------------
// Flash attention: one wave = one 16-query tile of one (b,h).
// Q,K: [B,S,NH,HD] bf16 (Q pre-scaled by 1/sqrt(HD)); V: [B,NH,HD,S] bf16.
// Out: [B,S,NH*HD] bf16.
// ---------------------------------------------------------------------------
__global__ __launch_bounds__(128) void mla_flash_attn(
    const __bf16* __restrict__ Q, const __bf16* __restrict__ Kc,
    const __bf16* __restrict__ Vt, __bf16* __restrict__ O) {
  __shared__ __bf16 ldsP[4][16][32];    // per-wave P tile (wave-private)

  const int lane = threadIdx.x & 31;
  const int wv   = threadIdx.x >> 5;
  const int gw   = blockIdx.x * 4 + wv;
  const int qTiles = S_ / 16;           // 128
  const int qt = gw % qTiles;
  const int h  = (gw / qTiles) % NH_;
  const int b  = gw / (qTiles * NH_);
  const int q0 = qt * 16;
  const int half = lane >> 4;
  const int l16  = lane & 15;

  // Q fragments: 4 K-chunks of 32 over HD=128
  v16bf qf[4];
  {
    const long qbase = ((long)(b * S_ + q0 + l16) * NH_ + h) * HD_;
#pragma unroll
    for (int c = 0; c < 4; ++c) {
      const __bf16* p = Q + qbase + c * 32 + 8 * half;
      qf[c] = combine16(*(const v8bf*)p, *(const v8bf*)(p + 16));
    }
  }

  float mrow[8], lrow[8];
  v8f oacc[8];
#pragma unroll
  for (int j = 0; j < 8; ++j) {
    mrow[j] = -3.0e38f; lrow[j] = 0.0f;
#pragma unroll
    for (int nt = 0; nt < 8; ++nt) oacc[nt][j] = 0.0f;
  }

  const int kend = q0 + 16;             // causal: keys < q0+16
  for (int k0 = 0; k0 < kend; k0 += 32) {
    // ---- S = Q K^T : two 16x16 score tiles -------------------------------
    v8f sc[2];
#pragma unroll
    for (int kt = 0; kt < 2; ++kt) {
      v8f a;
#pragma unroll
      for (int j = 0; j < 8; ++j) a[j] = 0.0f;
      const int key = k0 + kt * 16 + l16;
      const long kbase = ((long)(b * S_ + key) * NH_ + h) * HD_;
#pragma unroll
      for (int c = 0; c < 4; ++c) {
        const __bf16* p = Kc + kbase + c * 32 + 8 * half;
        v16bf bfr = combine16(*(const v8bf*)p, *(const v8bf*)(p + 16));
        a = WMMA_BF16(qf[c], bfr, a);
      }
      sc[kt] = a;
    }

    // ---- causal mask + online softmax ------------------------------------
#pragma unroll
    for (int j = 0; j < 8; ++j) {
      const int row = q0 + j + 8 * half;
      float v0 = (k0 + l16      <= row) ? sc[0][j] : -3.0e38f;
      float v1 = (k0 + 16 + l16 <= row) ? sc[1][j] : -3.0e38f;
      float mx = fmaxf(v0, v1);
#pragma unroll
      for (int off = 1; off < 16; off <<= 1)
        mx = fmaxf(mx, __shfl_xor(mx, off, 32));
      float newm = fmaxf(mrow[j], mx);
      float corr = __expf(mrow[j] - newm);
      mrow[j] = newm;
      float p0 = __expf(v0 - newm);
      float p1 = __expf(v1 - newm);
      sc[0][j] = p0; sc[1][j] = p1;
      float rs = p0 + p1;
#pragma unroll
      for (int off = 1; off < 16; off <<= 1)
        rs += __shfl_xor(rs, off, 32);
      lrow[j] = lrow[j] * corr + rs;
#pragma unroll
      for (int nt = 0; nt < 8; ++nt) oacc[nt][j] *= corr;
    }

    // ---- transpose P via LDS (C-layout -> A-fragment layout) -------------
#pragma unroll
    for (int j = 0; j < 8; ++j) {
      const int r = j + 8 * half;
      ldsP[wv][r][l16]      = (__bf16)sc[0][j];
      ldsP[wv][r][16 + l16] = (__bf16)sc[1][j];
    }
    // LDS ops are in-order within a wave; no barrier needed for wave-private tile.
    v16bf pf;
    {
      const __bf16* p = &ldsP[wv][l16][8 * half];
      pf = combine16(*(const v8bf*)p, *(const v8bf*)(p + 16));
    }

    // ---- O += P V : V pre-transposed [b,h,d,s] so loads are contiguous ---
#pragma unroll
    for (int nt = 0; nt < 8; ++nt) {
      const __bf16* p = Vt + ((long)(b * NH_ + h) * HD_ + nt * 16 + l16) * S_ +
                        k0 + 8 * half;
      v16bf vf = combine16(*(const v8bf*)p, *(const v8bf*)(p + 16));
      oacc[nt] = WMMA_BF16(pf, vf, oacc[nt]);
    }
  }

  // ---- normalize + store [b, q, h*HD + d] --------------------------------
#pragma unroll
  for (int j = 0; j < 8; ++j) {
    const float inv = 1.0f / lrow[j];
    const int row = q0 + j + 8 * half;
    const long base = (long)(b * S_ + row) * H_ + h * HD_;
#pragma unroll
    for (int nt = 0; nt < 8; ++nt)
      O[base + nt * 16 + l16] = (__bf16)(oacc[nt][j] * inv);
  }
}

// ---------------------------------------------------------------------------
// Host launcher
// ---------------------------------------------------------------------------
static inline __bf16* ws_bf(void* ws, size_t& off, size_t elems) {
  __bf16* p = (__bf16*)((char*)ws + off);
  off += ((elems * sizeof(__bf16)) + 255) & ~(size_t)255;
  return p;
}

extern "C" void kernel_launch(void* const* d_in, const int* in_sizes, int n_in,
                              void* d_out, int out_size, void* d_ws,
                              size_t ws_size, hipStream_t stream) {
  (void)in_sizes; (void)n_in; (void)out_size; (void)ws_size;
  const float* hidden = (const float*)d_in[0];
  // d_in[1] = attention_mask: pure causal mask, applied analytically in-kernel.
  const float* cosp = (const float*)d_in[2];
  const float* sinp = (const float*)d_in[3];
  const float* Wq   = (const float*)d_in[4];
  const float* Wkc  = (const float*)d_in[5];
  const float* Wvc  = (const float*)d_in[6];
  const float* Wkd  = (const float*)d_in[7];
  const float* Wvd  = (const float*)d_in[8];
  const float* Wo   = (const float*)d_in[9];

  size_t off = 0;
  __bf16* Xbf  = ws_bf(d_ws, off, (size_t)TOK_ * H_);
  __bf16* Wqb  = ws_bf(d_ws, off, (size_t)H_ * H_);
  __bf16* Wkcb = ws_bf(d_ws, off, (size_t)LAT_ * H_);
  __bf16* Wvcb = ws_bf(d_ws, off, (size_t)LAT_ * H_);
  __bf16* Wkdb = ws_bf(d_ws, off, (size_t)H_ * LAT_);
  __bf16* Wvdb = ws_bf(d_ws, off, (size_t)H_ * LAT_);
  __bf16* Wob  = ws_bf(d_ws, off, (size_t)H_ * H_);
  __bf16* klat = ws_bf(d_ws, off, (size_t)TOK_ * LAT_);
  __bf16* vlat = ws_bf(d_ws, off, (size_t)TOK_ * LAT_);
  __bf16* qtmp = ws_bf(d_ws, off, (size_t)TOK_ * H_);
  __bf16* ktmp = ws_bf(d_ws, off, (size_t)TOK_ * H_);
  __bf16* qro  = ws_bf(d_ws, off, (size_t)TOK_ * H_);
  __bf16* kro  = ws_bf(d_ws, off, (size_t)TOK_ * H_);
  __bf16* vt   = ws_bf(d_ws, off, (size_t)TOK_ * H_);
  __bf16* attn = qtmp;  // reuse: qtmp is dead after RoPE

  auto cvt = [&](const float* s, __bf16* d, long n) {
    cvt_f32_bf16<<<1024, 256, 0, stream>>>(s, d, n / 4);
  };
  cvt(hidden, Xbf, (long)TOK_ * H_);
  cvt(Wq,  Wqb,  (long)H_ * H_);
  cvt(Wkc, Wkcb, (long)LAT_ * H_);
  cvt(Wvc, Wvcb, (long)LAT_ * H_);
  cvt(Wkd, Wkdb, (long)H_ * LAT_);
  cvt(Wvd, Wvdb, (long)H_ * LAT_);
  cvt(Wo,  Wob,  (long)H_ * H_);

  auto gemm_blocks = [](int M, int N) { return ((M / 32) * (N / 64)) / 8; };

  // Projections (bf16 out)
  gemm_bf16<1><<<gemm_blocks(TOK_, H_),   256, 0, stream>>>(Xbf, Wqb,  qtmp, TOK_, H_,   H_);
  gemm_bf16<1><<<gemm_blocks(TOK_, LAT_), 256, 0, stream>>>(Xbf, Wkcb, klat, TOK_, LAT_, H_);
  gemm_bf16<1><<<gemm_blocks(TOK_, LAT_), 256, 0, stream>>>(Xbf, Wvcb, vlat, TOK_, LAT_, H_);
  gemm_bf16<1><<<gemm_blocks(TOK_, H_),   256, 0, stream>>>(klat, Wkdb, ktmp, TOK_, H_, LAT_);
  gemm_bf16<2><<<gemm_blocks(TOK_, H_),   256, 0, stream>>>(vlat, Wvdb, vt,   TOK_, H_, LAT_);

  // RoPE (fold 1/sqrt(HD) into Q)
  const float qscale = 0.08838834764831845f;  // 1/sqrt(128)
  rope_kernel<<<2048, 256, 0, stream>>>(qtmp, cosp, sinp, qro, qscale);
  rope_kernel<<<2048, 256, 0, stream>>>(ktmp, cosp, sinp, kro, 1.0f);

  // Flash attention: B*NH*(S/16) = 8192 waves, 4 waves/block
  mla_flash_attn<<<(B_ * NH_ * (S_ / 16)) / 4, 128, 0, stream>>>(qro, kro, vt, attn);

  // Output projection -> fp32 d_out
  gemm_bf16<0><<<gemm_blocks(TOK_, H_), 256, 0, stream>>>(attn, Wob, (float*)d_out,
                                                          TOK_, H_, H_);
}
